// RENTransitionModel_30305289240672
// MI455X (gfx1250) — compile-verified, matrix operand order checked by reading, not run on hardware
//
#include <hip/hip_runtime.h>
#include <cmath>

// ---------------------------------------------------------------------------
// RENTransitionModel for MI455X (gfx1250, wave32, WMMA).
//
// Kernel 1 (ren_setup, 1 block): builds M = H^T H + eps*I, inverts
// E = g^2 I - M22 by Gauss-Jordan, forms A/B1/B2, then packs ALL weight
// matrices as bf16 WMMA B-fragments ([frag][lane][16 halves]) into d_ws.
//
// Kernel 2 (ren_main, 2048 blocks x 256 thr = 8 waves x 16 rows): copies the
// 94 KB fragment buffer into LDS once, then each wave runs the whole row-tile
// pipeline in WMMA f32 accumulators, using a per-wave LDS staging buffer to
// re-layout tanh(v)/zt1 from C/D-fragment layout into A-fragment layout.
// ---------------------------------------------------------------------------

typedef __attribute__((ext_vector_type(16))) __bf16 v16bf;
typedef __attribute__((ext_vector_type(8)))  float  v8f;
typedef __attribute__((ext_vector_type(4)))  float  f4;

#define N_Y     20
#define N_FP    5
#define GAMMA2  0.9801f
#define REN_EPS 0.01f

// workspace layout (float units)
#define WS_M    0                 // 224*224
#define WS_AUG  50176             // 128*256 (Gauss-Jordan augmented [E|I])
#define WS_A    82944             // 128*128
#define WS_B1   99328             // 128*64
#define WS_B2   107520            // 128*32
#define WS_FRAG 111616            // 94*512 bf16 fragment buffer

#define NFRAG   94
// fragment index bases (each fragment = 512 bf16 = 32 lanes x 16 halves)
#define FB_C1T  0    // C1^T : 4 n-tiles x 4 k-chunks
#define FB_D12T 16   // D12^T: 4 x 1
#define FB_D11T 20   // D11^T: 4 x 2
#define FB_AT   28   // A^T  : 8 x 4
#define FB_B1T  60   // B1^T : 8 x 2
#define FB_B2T  76   // B2^T : 8 x 1
#define FB_CT   84   // C^T  : 2 x 4 (N_Y padded to 32)
#define FB_DT   92   // D^T  : 2 x 1

// ---------------------------------------------------------------------------
// setup kernel
// ---------------------------------------------------------------------------
__global__ __launch_bounds__(256) void ren_setup(
    const float* __restrict__ H, const float* __restrict__ Cin,
    const float* __restrict__ Din, float* __restrict__ ws)
{
  const int tid = threadIdx.x;
  float* M   = ws + WS_M;
  float* aug = ws + WS_AUG;
  float* Aw  = ws + WS_A;
  float* B1w = ws + WS_B1;
  float* B2w = ws + WS_B2;
  __shared__ float fac[128];

  // M = H^T H + eps I  (224x224, H is 64x224)
  for (int i = tid; i < 224*224; i += 256) {
    int r = i / 224, c = i % 224;
    float s = 0.f;
    for (int q = 0; q < 64; ++q) s += H[q*224 + r] * H[q*224 + c];
    if (r == c) s += REN_EPS;
    M[i] = s;
  }
  __syncthreads();

  // aug = [E | I], E = g2*I - M22
  for (int i = tid; i < 128*256; i += 256) {
    int r = i >> 8, c = i & 255;
    float v;
    if (c < 128) v = (c == r ? GAMMA2 : 0.f) - M[(64+r)*224 + 64 + c];
    else         v = ((c - 128) == r) ? 1.f : 0.f;
    aug[i] = v;
  }

  // Gauss-Jordan: 256 threads, one column each
  for (int p = 0; p < 128; ++p) {
    __syncthreads();
    float piv = aug[p*256 + p];
    __syncthreads();
    aug[p*256 + tid] *= (1.f / piv);
    __syncthreads();
    if (tid < 128) fac[tid] = (tid == p) ? 0.f : aug[tid*256 + p];
    __syncthreads();
    float pr = aug[p*256 + tid];
    for (int r = 0; r < 128; ++r) aug[r*256 + tid] -= fac[r] * pr;
  }
  __syncthreads();

  // A = 0.5 * Einv @ (g2 I + M22);  Einv[i][k] = aug[i*256+128+k]
  for (int i = tid; i < 128*128; i += 256) {
    int r = i >> 7, c = i & 127;
    float s = 0.f;
    for (int k = 0; k < 128; ++k) {
      float rhs = (k == c ? GAMMA2 : 0.f) + M[(64+k)*224 + 64 + c];
      s += aug[r*256 + 128 + k] * rhs;
    }
    Aw[i] = 0.5f * s;
  }
  // B1 = Einv @ M12^T   (M12[j][k] = M[j*224 + 64 + k], j<64)
  for (int i = tid; i < 128*64; i += 256) {
    int r = i >> 6, c = i & 63;
    float s = 0.f;
    for (int k = 0; k < 128; ++k) s += aug[r*256 + 128 + k] * M[c*224 + 64 + k];
    B1w[i] = s;
  }
  // B2 = Einv @ M23     (M23[k][j] = M[(64+k)*224 + 192 + j])
  for (int i = tid; i < 128*32; i += 256) {
    int r = i >> 5, c = i & 31;
    float s = 0.f;
    for (int k = 0; k < 128; ++k) s += aug[r*256 + 128 + k] * M[(64+k)*224 + 192 + c];
    B2w[i] = s;
  }
  __syncthreads();

  // pack bf16 B-fragments: frag[lane][h] = W[t*16 + (lane&15)][kc*32 + khalf]
  // khalf = h + 8*(h>>3) + 8*(lane>>4)   (16-bit A/B VGPR layout, ISA 7.12.2)
  __bf16* fb = (__bf16*)(ws + WS_FRAG);
  for (int e = tid; e < NFRAG*512; e += 256) {
    int frag = e >> 9;
    int lane = (e >> 4) & 31;
    int h    = e & 15;
    int kh   = h + 8*(h >> 3) + 8*(lane >> 4);
    int nloc = lane & 15;
    float val;
    if (frag < FB_D12T)      { int f = frag;          int n = (f>>2)*16+nloc, K = (f&3)*32+kh; val = M[n*224 + 64 + K]; }
    else if (frag < FB_D11T) { int f = frag-FB_D12T;  int n = f*16+nloc,      K = kh;          val = M[n*224 + 192 + K]; }
    else if (frag < FB_AT)   { int f = frag-FB_D11T;  int n = (f>>1)*16+nloc, K = (f&1)*32+kh; val = M[n*224 + K]; }
    else if (frag < FB_B1T)  { int f = frag-FB_AT;    int n = (f>>2)*16+nloc, K = (f&3)*32+kh; val = Aw[n*128 + K]; }
    else if (frag < FB_B2T)  { int f = frag-FB_B1T;   int n = (f>>1)*16+nloc, K = (f&1)*32+kh; val = B1w[n*64 + K]; }
    else if (frag < FB_CT)   { int f = frag-FB_B2T;   int n = f*16+nloc,      K = kh;          val = B2w[n*32 + K]; }
    else if (frag < FB_DT)   { int f = frag-FB_CT;    int n = (f>>2)*16+nloc, K = (f&3)*32+kh; val = (n < N_Y) ? Cin[n*128 + K] : 0.f; }
    else                     { int f = frag-FB_DT;    int n = f*16+nloc,      K = kh;          val = (n < N_Y) ? Din[n*32 + K]  : 0.f; }
    fb[e] = (__bf16)val;
  }
}

// ---------------------------------------------------------------------------
// main kernel helpers
// ---------------------------------------------------------------------------
__device__ __forceinline__ float gtanh(float x) {
#if __has_builtin(__builtin_amdgcn_tanhf)
  return __builtin_amdgcn_tanhf(x);
#else
  return tanhf(x);
#endif
}

__device__ __forceinline__ v8f wmma_bf16(v16bf a, v16bf b, v8f c) {
  return __builtin_amdgcn_wmma_f32_16x16x32_bf16(false, a, false, b, (short)0, c, false, false);
}

// load an A-fragment (16 rows x K=32 chunk) from row-major f32, scaled, ->bf16
__device__ __forceinline__ v16bf load_a_frag(const float* __restrict__ base,
                                             int row, int ld, int kbase,
                                             int lane, float scale) {
  const float* p = base + (size_t)row*ld + kbase + 8*(lane >> 4);
  f4 a0 = *(const f4*)(p + 0);
  f4 a1 = *(const f4*)(p + 4);
  f4 b0 = *(const f4*)(p + 16);
  f4 b1 = *(const f4*)(p + 20);
  v16bf r;
  r[0]=(__bf16)(a0.x*scale); r[1]=(__bf16)(a0.y*scale); r[2]=(__bf16)(a0.z*scale); r[3]=(__bf16)(a0.w*scale);
  r[4]=(__bf16)(a1.x*scale); r[5]=(__bf16)(a1.y*scale); r[6]=(__bf16)(a1.z*scale); r[7]=(__bf16)(a1.w*scale);
  r[8]=(__bf16)(b0.x*scale); r[9]=(__bf16)(b0.y*scale); r[10]=(__bf16)(b0.z*scale); r[11]=(__bf16)(b0.w*scale);
  r[12]=(__bf16)(b1.x*scale); r[13]=(__bf16)(b1.y*scale); r[14]=(__bf16)(b1.z*scale); r[15]=(__bf16)(b1.w*scale);
  return r;
}

// C/D fragments (NT n-tiles) -> bf16 A-fragments via per-wave LDS staging.
// Same-wave LDS ops are in-order; one s_wait_dscnt is enough (no barrier).
template<int NT, bool TANH>
__device__ __forceinline__ void relayout_to_afrag(const v8f* acc, __bf16* st,
                                                  int lane, v16bf* out) {
  const int laneBase = (8*(lane >> 4) + 16*((lane >> 3) & 1))*16 + (lane & 7);
  #pragma unroll
  for (int t = 0; t < NT; ++t) {
    v8f a = acc[t];
    #pragma unroll
    for (int r = 0; r < 8; ++r) {
      float x = a[r];
      if (TANH) x = gtanh(x);
      st[laneBase + (t >> 1)*512 + r*16 + (t & 1)*8] = (__bf16)x;
    }
  }
  asm volatile("s_wait_dscnt 0x0" ::: "memory");
  #pragma unroll
  for (int kc = 0; kc < NT/2; ++kc)
    out[kc] = *(const v16bf*)(st + kc*512 + lane*16);
  asm volatile("" ::: "memory");
}

// ---------------------------------------------------------------------------
// main kernel: 256 threads = 8 waves, 16 rows per wave
// ---------------------------------------------------------------------------
__global__ __launch_bounds__(256) void ren_main(
    const float* __restrict__ zt, const float* __restrict__ dtp,
    const float* __restrict__ ut, const float* __restrict__ bv,
    const float* __restrict__ bx, const float* __restrict__ ws,
    float* __restrict__ out_z, float* __restrict__ out_y)
{
  __shared__ __align__(16) __bf16 lds_w[NFRAG*512];   // 96256 B
  __shared__ float lds_bv[64];
  __shared__ float lds_bx[128];
  __shared__ __align__(32) __bf16 lds_stage[8*2048];  // 4 KB per wave

  const int tid = threadIdx.x;
  {
    const uint4* src = (const uint4*)(ws + WS_FRAG);
    uint4* dst = (uint4*)lds_w;
    for (int i = tid; i < NFRAG*512/8; i += 256) dst[i] = src[i];
    if (tid < 64)  lds_bv[tid] = bv[tid];
    if (tid < 128) lds_bx[tid] = bx[tid];
  }
  __syncthreads();

  const int lane = tid & 31;
  const int wave = tid >> 5;
  const int rowBase = blockIdx.x*128 + wave*16;
  const int myRow = rowBase + (lane & 15);
  const float dtv = dtp[0];
  __bf16* st = lds_stage + wave*2048;

  #define WF(idx) (*(const v16bf*)(lds_w + (idx)*512 + lane*16))

  // activation A-fragments
  v16bf zA[4];
  #pragma unroll
  for (int kc = 0; kc < 4; ++kc) zA[kc] = load_a_frag(zt, myRow, 128, kc*32, lane, 1.f);
  v16bf uA = load_a_frag(ut, myRow, 32, 0, lane, dtv);   // ut * dt fused

  // base = zt@C1^T + ut_dt@D12^T + bv
  v8f baseAcc[4], vAcc[4];
  #pragma unroll
  for (int t = 0; t < 4; ++t) {
    float b = lds_bv[t*16 + (lane & 15)];
    v8f acc = {b,b,b,b,b,b,b,b};
    #pragma unroll
    for (int kc = 0; kc < 4; ++kc) acc = wmma_bf16(zA[kc], WF(FB_C1T + t*4 + kc), acc);
    acc = wmma_bf16(uA, WF(FB_D12T + t), acc);
    baseAcc[t] = acc;
    vAcc[t] = acc;
  }

  // fixed point: v = base + tanh(v)@D11^T  (x N_FP)
  v16bf wt[2];
  #pragma unroll
  for (int it = 0; it < N_FP; ++it) {
    relayout_to_afrag<4, true>(vAcc, st, lane, wt);
    #pragma unroll
    for (int t = 0; t < 4; ++t) {
      v8f acc = baseAcc[t];
      acc = wmma_bf16(wt[0], WF(FB_D11T + t*2 + 0), acc);
      acc = wmma_bf16(wt[1], WF(FB_D11T + t*2 + 1), acc);
      vAcc[t] = acc;
    }
  }
  relayout_to_afrag<4, true>(vAcc, st, lane, wt);  // w = tanh(v)

  // zt1 = zt@A^T + w@B1^T + ut_dt@B2^T + bx
  v8f z1[8];
  #pragma unroll
  for (int t = 0; t < 8; ++t) {
    float b = lds_bx[t*16 + (lane & 15)];
    v8f acc = {b,b,b,b,b,b,b,b};
    #pragma unroll
    for (int kc = 0; kc < 4; ++kc) acc = wmma_bf16(zA[kc], WF(FB_AT + t*4 + kc), acc);
    acc = wmma_bf16(wt[0], WF(FB_B1T + t*2 + 0), acc);
    acc = wmma_bf16(wt[1], WF(FB_B1T + t*2 + 1), acc);
    acc = wmma_bf16(uA, WF(FB_B2T + t), acc);
    z1[t] = acc;
  }

  // store zt1 (f32, column-per-lane coalesced in 64B segments)
  const int mrow = rowBase + 8*(lane >> 4);
  #pragma unroll
  for (int t = 0; t < 8; ++t) {
    v8f a = z1[t];
    #pragma unroll
    for (int r = 0; r < 8; ++r)
      out_z[(size_t)(mrow + r)*128 + t*16 + (lane & 15)] = a[r];
  }

  // yt1 = zt1@C^T + ut_dt@D^T  (N_Y padded to 32)
  v16bf z1A[4];
  relayout_to_afrag<8, false>(z1, st, lane, z1A);
  #pragma unroll
  for (int t = 0; t < 2; ++t) {
    v8f acc = {0,0,0,0,0,0,0,0};
    #pragma unroll
    for (int kc = 0; kc < 4; ++kc) acc = wmma_bf16(z1A[kc], WF(FB_CT + t*4 + kc), acc);
    acc = wmma_bf16(uA, WF(FB_DT + t), acc);
    int col = t*16 + (lane & 15);
    if (col < N_Y) {
      #pragma unroll
      for (int r = 0; r < 8; ++r)
        out_y[(size_t)(mrow + r)*N_Y + col] = acc[r];
    }
  }
  #undef WF
}

// ---------------------------------------------------------------------------
extern "C" void kernel_launch(void* const* d_in, const int* in_sizes, int n_in,
                              void* d_out, int out_size, void* d_ws, size_t ws_size,
                              hipStream_t stream) {
  const float* zt = (const float*)d_in[0];
  const float* dt = (const float*)d_in[1];
  const float* ut = (const float*)d_in[2];
  const float* H  = (const float*)d_in[3];
  const float* bv = (const float*)d_in[4];
  const float* bx = (const float*)d_in[5];
  const float* C  = (const float*)d_in[6];
  const float* D  = (const float*)d_in[7];
  float* ws = (float*)d_ws;           // needs ~543 KB
  int Btot = in_sizes[0] / 128;       // 262144
  float* outz = (float*)d_out;
  float* outy = outz + (size_t)Btot * 128;

  ren_setup<<<1, 256, 0, stream>>>(H, C, D, ws);
  ren_main<<<Btot / 128, 256, 0, stream>>>(zt, dt, ut, bv, bx, ws, outz, outy);
}
